// DivisionIteration_23407571764115
// MI455X (gfx1250) — matrix operand order repaired and checked by reading, not compiled, not run on hardware
//
#include <hip/hip_runtime.h>

// CDNA5 / gfx1250 (wave32) restoring-division iteration (48-bit).
// Bit-packing of {0,1} float bit-vectors is done EXACTLY with
// v_wmma_f32_16x16x32_bf16 (powers of two are exact in bf16; f32
// accumulation of <=24 distinct powers of two stays < 2^24, exact).
// Kernel is memory-bound (~580 MB total traffic -> ~25 us at 23.3 TB/s),
// so bulk traffic uses non-temporal hints and b128 accesses.

typedef __attribute__((ext_vector_type(16))) __bf16       v16bf;
typedef __attribute__((ext_vector_type(8)))  float        v8f;
typedef __attribute__((ext_vector_type(8)))  unsigned int v8u;
typedef __attribute__((ext_vector_type(4)))  float        v4f;
typedef __attribute__((ext_vector_type(4)))  unsigned int v4u;

// bf16(f0) -> [15:0], bf16(f1) -> [31:16]; exact for f in {0.0f, 1.0f}
// (bf16 bits of an f32 that is 0.0/1.0 are just its top 16 bits).
__device__ __forceinline__ unsigned pack_bf16pair(float f0, float f1) {
  // v_perm_b32: src1 supplies byte indices 0..3, src0 supplies 4..7.
  return __builtin_amdgcn_perm(__float_as_uint(f1), __float_as_uint(f0),
                               0x07060302u);
}

// bit pattern of bf16(2^e)
__device__ __forceinline__ unsigned bfexp(int e) {
  return (unsigned)((127 + e) << 7);
}

__device__ __forceinline__ v8f wmma_bf16(v8u a, v8u b, v8f c) {
  return __builtin_amdgcn_wmma_f32_16x16x32_bf16(
      /*neg_a=*/false, __builtin_bit_cast(v16bf, a),
      /*neg_b=*/false, __builtin_bit_cast(v16bf, b),
      /*c_mod=*/(short)0, c, /*reuse_a=*/false, /*reuse_b=*/false);
}

__device__ __forceinline__ v4f ntload4(const float* p) {
  return __builtin_nontemporal_load((const v4f*)p);
}

__global__ __launch_bounds__(256) void division_iteration_wmma_kernel(
    const float* __restrict__ R, const float* __restrict__ D,
    float* __restrict__ out, int nrows) {
  __shared__ unsigned smem[8][32];   // per wave: 16 rows x {lo24, hi24}

  const int lane = threadIdx.x & 31;
  const int wv   = threadIdx.x >> 5;
  const int h    = lane >> 4;        // lane group (K-half selector)
  const int N    = lane & 15;        // A-matrix row M / B,C,D column N

  const int ntiles = (nrows + 15) >> 4;
  const int tile   = blockIdx.x * 8 + wv;   // one 16-row tile per wave
  if (tile >= ntiles) return;               // uniform per wave: EXEC stays full
  const int m0 = tile << 4;
  const bool fulltile = (m0 + 16 <= nrows); // wave-uniform

  // ---- load this lane's bit chunks (row M = N, columns depend on h) ----
  int row  = m0 + N;
  int lrow = row < nrows ? row : nrows - 1;           // clamp (safe reads)
  const float* rr = R + (size_t)lrow * 48 + 8 * h;
  const float* dd = D + (size_t)lrow * 48 + 8 * h;

  v4f ra0 = ntload4(rr);      v4f ra1 = ntload4(rr + 4);
  v4f rb0 = ntload4(rr + 16); v4f rb1 = ntload4(rr + 20);
  v4f rc0 = ntload4(rr + 32); v4f rc1 = ntload4(rr + 36);
  v4f da0 = ntload4(dd);      v4f da1 = ntload4(dd + 4);
  v4f db0 = ntload4(dd + 16); v4f db1 = ntload4(dd + 20);
  v4f dc0 = ntload4(dd + 32); v4f dc1 = ntload4(dd + 36);

  // ---- pack into A-matrix (16-bit A 16x32 layout) ----
  // VGPR v in 0..3: K = 8h + 2v ; VGPR v in 4..7: K = 16 + 8h + 2(v-4)
  v8u aR1, aR2, aD1, aD2;
  aR1[0] = pack_bf16pair(ra0.x, ra0.y); aR1[1] = pack_bf16pair(ra0.z, ra0.w);
  aR1[2] = pack_bf16pair(ra1.x, ra1.y); aR1[3] = pack_bf16pair(ra1.z, ra1.w);
  aR1[4] = pack_bf16pair(rb0.x, rb0.y); aR1[5] = pack_bf16pair(rb0.z, rb0.w);
  aR1[6] = pack_bf16pair(rb1.x, rb1.y); aR1[7] = pack_bf16pair(rb1.z, rb1.w);
  aR2[0] = pack_bf16pair(rc0.x, rc0.y); aR2[1] = pack_bf16pair(rc0.z, rc0.w);
  aR2[2] = pack_bf16pair(rc1.x, rc1.y); aR2[3] = pack_bf16pair(rc1.z, rc1.w);
  aR2[4] = 0u; aR2[5] = 0u; aR2[6] = 0u; aR2[7] = 0u;

  aD1[0] = pack_bf16pair(da0.x, da0.y); aD1[1] = pack_bf16pair(da0.z, da0.w);
  aD1[2] = pack_bf16pair(da1.x, da1.y); aD1[3] = pack_bf16pair(da1.z, da1.w);
  aD1[4] = pack_bf16pair(db0.x, db0.y); aD1[5] = pack_bf16pair(db0.z, db0.w);
  aD1[6] = pack_bf16pair(db1.x, db1.y); aD1[7] = pack_bf16pair(db1.z, db1.w);
  aD2[0] = pack_bf16pair(dc0.x, dc0.y); aD2[1] = pack_bf16pair(dc0.z, dc0.w);
  aD2[2] = pack_bf16pair(dc1.x, dc1.y); aD2[3] = pack_bf16pair(dc1.z, dc1.w);
  aD2[4] = 0u; aD2[5] = 0u; aD2[6] = 0u; aD2[7] = 0u;

  // ---- constant weight B matrices (16-bit B 32x16 layout: K = 8h + 2v (v<4), 16 + 8h + 2(v-4) (v>=4)) ----
  // cols N<8 : lo24 weights (2^K for K<24);  cols N>=8: hi24 weights.
  v8u b1, b2;
#pragma unroll
  for (int v = 0; v < 8; ++v) {
    int K0 = (v < 4) ? 8 * h + 2 * v : 16 + 8 * h + 2 * (v - 4), K1 = K0 + 1;
    unsigned w1 = 0u, w2 = 0u;
    if (N < 8) {
      unsigned x0 = (K0 < 24) ? bfexp(K0) : 0u;
      unsigned x1 = (K1 < 24) ? bfexp(K1) : 0u;
      w1 = x0 | (x1 << 16);
    } else {
      unsigned x0 = (K0 >= 24) ? bfexp(K0 - 24) : 0u;
      unsigned x1 = (K1 >= 24) ? bfexp(K1 - 24) : 0u;
      w1 = x0 | (x1 << 16);
      unsigned y0 = (K0 < 16) ? bfexp(K0 + 8) : 0u;   // bits 32.. -> 2^(K'+8)
      unsigned y1 = (K1 < 16) ? bfexp(K1 + 8) : 0u;
      w2 = y0 | (y1 << 16);
    }
    b1[v] = w1; b2[v] = w2;
  }

  // ---- exact bit packing via WMMA: D(M, N<8)=lo24(M), D(M, N>=8)=hi24(M) ----
  v8f accR = {}; v8f accD = {};
  accR = wmma_bf16(aR1, b1, accR);
  accR = wmma_bf16(aR2, b2, accR);
  accD = wmma_bf16(aD1, b1, accD);
  accD = wmma_bf16(aD2, b2, accD);

  // ---- per-row 48-bit subtract-with-restore (exact in f32: all ints < 2^24)
  // C/D layout: VGPR v holds row M = v + 8h; lanes 0..7 lo, 8..15 hi.
  unsigned lo_a[8], hi_a[8];
  float    qf_a[8];
#pragma unroll
  for (int v = 0; v < 8; ++v) {
    float ownR = accR[v], ownD = accD[v];
    // XOR-8 swizzle (SWAPX8): exchange with the lo/hi partner lane
    float parR = __uint_as_float(
        (unsigned)__builtin_amdgcn_ds_swizzle((int)__float_as_uint(ownR), 0x201F));
    float parD = __uint_as_float(
        (unsigned)__builtin_amdgcn_ds_swizzle((int)__float_as_uint(ownD), 0x201F));
    bool lowh = (N < 8);
    float loR = lowh ? ownR : parR, hiR = lowh ? parR : ownR;
    float loD = lowh ? ownD : parD, hiD = lowh ? parD : ownD;

    float dl  = loR - loD;                       // exact, |.| < 2^24
    float bor = (dl < 0.0f) ? 1.0f : 0.0f;
    dl += bor * 16777216.0f;                     // + 2^24 on borrow
    float dh = (hiR - hiD) - bor;                // exact
    bool  q  = (dh >= 0.0f);                     // Q = (R >= D) = carry_out
    lo_a[v] = (unsigned)(q ? dl : loR);
    hi_a[v] = (unsigned)(q ? dh : hiR);
    qf_a[v] = q ? 1.0f : 0.0f;
  }

  // ---- designated lanes (N==0, i.e. lanes 0 and 16) publish results ----
  if (N == 0) {
    if (fulltile) {
      // 16 contiguous u32 per half -> 4x ds_store_b128
#pragma unroll
      for (int j = 0; j < 4; ++j) {
        v4u pk = {lo_a[2 * j], hi_a[2 * j], lo_a[2 * j + 1], hi_a[2 * j + 1]};
        *(v4u*)&smem[wv][16 * h + 4 * j] = pk;
      }
      // 8 contiguous Q floats -> 2x b128 NT global stores
      v4f q0 = {qf_a[0], qf_a[1], qf_a[2], qf_a[3]};
      v4f q1 = {qf_a[4], qf_a[5], qf_a[6], qf_a[7]};
      __builtin_nontemporal_store(q0, (v4f*)(out + m0 + 8 * h));
      __builtin_nontemporal_store(q1, (v4f*)(out + m0 + 8 * h + 4));
    } else {
#pragma unroll
      for (int v = 0; v < 8; ++v) {
        int r = v + 8 * h;
        if (m0 + r < nrows) {
          smem[wv][2 * r + 0] = lo_a[v];
          smem[wv][2 * r + 1] = hi_a[v];
          out[m0 + r] = qf_a[v];
        }
      }
    }
  }
  __builtin_amdgcn_wave_barrier();               // same-wave LDS is in-order

  // ---- expand packed results back to 0/1 floats, coalesced b128 NT stores --
  const size_t outRbase = (size_t)nrows + (size_t)m0 * 48;
  if (fulltile) {
#pragma unroll
    for (int k = 0; k < 6; ++k) {
      int idx  = 4 * (lane + 32 * k);            // 0..764, 16B aligned
      int r    = idx / 48;
      int rem  = idx - 48 * r;
      int word = rem / 24;
      int sh   = rem - 24 * word;
      unsigned w = smem[wv][2 * r + word];
      v4f o;
      o.x = ((w >> (sh + 0)) & 1u) ? 1.0f : 0.0f;
      o.y = ((w >> (sh + 1)) & 1u) ? 1.0f : 0.0f;
      o.z = ((w >> (sh + 2)) & 1u) ? 1.0f : 0.0f;
      o.w = ((w >> (sh + 3)) & 1u) ? 1.0f : 0.0f;
      __builtin_nontemporal_store(o, (v4f*)(out + outRbase + idx));
    }
  } else {
#pragma unroll
    for (int k = 0; k < 6; ++k) {
      int idx  = 4 * (lane + 32 * k);
      int r    = idx / 48;
      if (m0 + r >= nrows) continue;
      int rem  = idx - 48 * r;
      int word = rem / 24;
      int sh   = rem - 24 * word;
      unsigned w = smem[wv][2 * r + word];
      v4f o;
      o.x = ((w >> (sh + 0)) & 1u) ? 1.0f : 0.0f;
      o.y = ((w >> (sh + 1)) & 1u) ? 1.0f : 0.0f;
      o.z = ((w >> (sh + 2)) & 1u) ? 1.0f : 0.0f;
      o.w = ((w >> (sh + 3)) & 1u) ? 1.0f : 0.0f;
      *(v4f*)(out + outRbase + idx) = o;
    }
  }
}

extern "C" void kernel_launch(void* const* d_in, const int* in_sizes, int n_in,
                              void* d_out, int out_size, void* d_ws, size_t ws_size,
                              hipStream_t stream) {
  const float* R = (const float*)d_in[0];
  const float* D = (const float*)d_in[1];
  float* out = (float*)d_out;
  const int nrows  = in_sizes[0] / 48;           // 1,000,000
  const int ntiles = (nrows + 15) / 16;
  const int blocks = (ntiles + 7) / 8;           // 8 waves (tiles) per block
  division_iteration_wmma_kernel<<<blocks, 256, 0, stream>>>(R, D, out, nrows);
}
